// CausalSelfAttention_15805479649908
// MI455X (gfx1250) — matrix operand order repaired and checked by reading, not compiled, not run on hardware
//
#include <hip/hip_runtime.h>
#include <hip/hip_bf16.h>

typedef __attribute__((ext_vector_type(16))) __bf16 v16bf;
typedef __attribute__((ext_vector_type(8)))  float  v8f;
typedef __attribute__((ext_vector_type(4)))  __bf16 bf16x4;

static constexpr int Bsz = 4;
static constexpr int T   = 2048;
static constexpr int C   = 1024;
static constexpr int H   = 16;
static constexpr int Dh  = 64;

// Build a v16bf fragment from two 16-byte LDS chunks (ds_load_b128 x2).
__device__ inline v16bf frag2(const __bf16* p0, const __bf16* p1) {
    union { v16bf v; uint4 q[2]; } u;
    u.q[0] = *(const uint4*)p0;
    u.q[1] = *(const uint4*)p1;
    return u.v;
}

// A-fragment (16x32 bf16): elems 0..7 -> K=8*hf.., elems 8..15 -> K=16+8*hf..
__device__ inline v16bf load_a(const __bf16* row, int hf) {
    return frag2(row + 8 * hf, row + 16 + 8 * hf);
}
// B-fragment (32x16 bf16): elems 0..7 -> K=8*hf.., elems 8..15 -> K=16+8*hf..
__device__ inline v16bf load_b(const __bf16* row, int hf) {
    return frag2(row + 8 * hf, row + 16 + 8 * hf);
}

__device__ inline v8f zero8() { v8f z = {0.f,0.f,0.f,0.f,0.f,0.f,0.f,0.f}; return z; }

// Async global->LDS copy of 16 bytes per lane (ASYNCcnt-tracked, CDNA5).
// VDST = VGPR holding the LDS byte address (low 32 bits of flat shared ptr),
// VADDR = 64-bit global address.
__device__ inline void async_copy_b128(void* lds_ptr, const void* gptr) {
    const unsigned int lds_addr = (unsigned int)(uintptr_t)lds_ptr;
    const unsigned long long gaddr = (unsigned long long)(uintptr_t)gptr;
    asm volatile("global_load_async_to_lds_b128 %0, %1, off"
                 :: "v"(lds_addr), "v"(gaddr) : "memory");
}
__device__ inline void wait_async0() {
    asm volatile("s_wait_asynccnt 0x0" ::: "memory");
}

#define WMMA_BF16(A, B, Cacc) \
    __builtin_amdgcn_wmma_f32_16x16x32_bf16(false, (A), false, (B), (short)0, (Cacc), false, false)

// ---------------------------------------------------------------------------
// Kernel 1: qkv = x @ w_qkv + b_qkv, scattered to Q/K/V in [B,H,T,D] bf16.
// M=8192, K=1024, N=3072. Block tile 128x128, K-step 32, 8 waves (4x2).
// ---------------------------------------------------------------------------
__global__ __launch_bounds__(256) void qkv_gemm(
    const float* __restrict__ X, const float* __restrict__ W,
    const float* __restrict__ bias,
    __bf16* __restrict__ Qo, __bf16* __restrict__ Ko, __bf16* __restrict__ Vo)
{
    constexpr int Kd = C;       // 1024
    constexpr int N  = 3 * C;   // 3072
    __shared__ __bf16 As[128 * 32];   // [m][k]
    __shared__ __bf16 Bs[128 * 32];   // [n][k] (transposed W tile)

    const int tid  = threadIdx.x;
    const int lane = tid & 31, wave = tid >> 5;
    const int ln   = lane & 15, hf = lane >> 4;
    const int wm   = wave & 3, wn = wave >> 2;
    const int m0   = blockIdx.y * 128;
    const int n0   = blockIdx.x * 128;

    v8f acc[2][4];
#pragma unroll
    for (int mi = 0; mi < 2; ++mi)
#pragma unroll
        for (int ni = 0; ni < 4; ++ni) acc[mi][ni] = zero8();

    const int xrow = tid >> 3, xcol = (tid & 7) * 4;
    const int bn = tid & 127, bkh = (tid >> 7) * 16;

    for (int k0 = 0; k0 < Kd; k0 += 32) {
        // --- stage X tile (128x32 f32 -> bf16, converts so no async path) ---
#pragma unroll
        for (int p = 0; p < 4; ++p) {
            const float4 xv = *(const float4*)(X + (size_t)(m0 + xrow + p * 32) * Kd + k0 + xcol);
            bf16x4 pk;
            pk[0] = (__bf16)xv.x; pk[1] = (__bf16)xv.y;
            pk[2] = (__bf16)xv.z; pk[3] = (__bf16)xv.w;
            *(bf16x4*)(&As[(xrow + p * 32) * 32 + xcol]) = pk;
        }
        // --- stage W tile (32x128 f32 -> bf16, transposed to [n][k]) ---
#pragma unroll
        for (int g = 0; g < 4; ++g) {
            bf16x4 pk;
#pragma unroll
            for (int j = 0; j < 4; ++j)
                pk[j] = (__bf16)W[(size_t)(k0 + bkh + g * 4 + j) * N + n0 + bn];
            *(bf16x4*)(&Bs[bn * 32 + bkh + g * 4]) = pk;
        }
        if (k0 + 32 < Kd) {
            __builtin_prefetch(X + (size_t)(m0 + xrow) * Kd + k0 + 32 + xcol, 0, 1);
            __builtin_prefetch(W + (size_t)(k0 + 32 + bkh) * N + n0 + bn, 0, 1);
        }
        __syncthreads();

        const v16bf a0 = load_a(&As[(wm * 32 +      ln) * 32], hf);
        const v16bf a1 = load_a(&As[(wm * 32 + 16 + ln) * 32], hf);
#pragma unroll
        for (int ni = 0; ni < 4; ++ni) {
            const v16bf b = load_b(&Bs[(wn * 64 + ni * 16 + ln) * 32], hf);
            acc[0][ni] = WMMA_BF16(a0, b, acc[0][ni]);
            acc[1][ni] = WMMA_BF16(a1, b, acc[1][ni]);
        }
        __syncthreads();
    }

    // Epilogue: bias add, scatter to Q/K/V [B,H,T,D] bf16.
#pragma unroll
    for (int ni = 0; ni < 4; ++ni) {
        const int j  = n0 + wn * 64 + ni * 16 + ln;
        const float bj = bias[j];
        const int which = j >> 10, cc = j & 1023;
        const int hh = cc >> 6, dd = cc & 63;
        __bf16* dst = (which == 0) ? Qo : (which == 1) ? Ko : Vo;
#pragma unroll
        for (int mi = 0; mi < 2; ++mi) {
#pragma unroll
            for (int r = 0; r < 8; ++r) {
                const int row = m0 + wm * 32 + mi * 16 + hf * 8 + r;
                const int bb = row >> 11, tt = row & (T - 1);
                dst[((size_t)(bb * H + hh) * T + tt) * Dh + dd] =
                    (__bf16)(acc[mi][ni][r] + bj);
            }
        }
    }
}

// ---------------------------------------------------------------------------
// Kernel 2: flash attention per (b, h, 128-query tile). Online softmax,
// S = Q K^T and O += P V via WMMA. KV tile = 64, causal.
// Q/K tiles staged via GLOBAL_LOAD_ASYNC_TO_LDS_B128 (ASYNCcnt).
// ---------------------------------------------------------------------------
__global__ __launch_bounds__(256) void flash_attn(
    const __bf16* __restrict__ Qg, const __bf16* __restrict__ Kg,
    const __bf16* __restrict__ Vg, __bf16* __restrict__ Og)
{
    __shared__ __bf16 Qs[128 * 64];      // [qrow][d]
    __shared__ __bf16 Ks[64 * 64];       // [key][d]
    __shared__ __bf16 Vt[64 * 64];       // [d][key] (transposed)
    __shared__ __bf16 Ps[8 * 16 * 64];   // per-wave P tile [16][64]

    const int tid  = threadIdx.x;
    const int lane = tid & 31, wave = tid >> 5;
    const int ln   = lane & 15, hf = lane >> 4;
    const int qt = blockIdx.x, h = blockIdx.y, b = blockIdx.z;
    const size_t headBase = (size_t)(b * H + h) * T * Dh;
    const float scale = 0.125f; // 1/sqrt(64)

    // Stage Q tile asynchronously (bf16 copy, 16B per lane per issue).
#pragma unroll
    for (int i = 0; i < 4; ++i) {
        const int idx = tid + i * 256;
        const int qr = idx >> 3, dg = (idx & 7) * 8;
        async_copy_b128(&Qs[qr * 64 + dg],
                        Qg + headBase + (size_t)(qt * 128 + qr) * Dh + dg);
    }

    float mI[8], lI[8];
    v8f o[4];
#pragma unroll
    for (int r = 0; r < 8; ++r) { mI[r] = -1e30f; lI[r] = 0.f; }
#pragma unroll
    for (int tn = 0; tn < 4; ++tn) o[tn] = zero8();

    const int nkv = 2 * qt + 2;
    for (int kvi = 0; kvi < nkv; ++kvi) {
        const int kv0 = kvi * 64;
        // Stage K tile [key][d] asynchronously.
#pragma unroll
        for (int i = 0; i < 2; ++i) {
            const int idx = tid + i * 256;
            const int kr = idx >> 3, dg = (idx & 7) * 8;
            async_copy_b128(&Ks[kr * 64 + dg],
                            Kg + headBase + (size_t)(kv0 + kr) * Dh + dg);
        }
        // Stage V transposed [d][key] (layout change -> through VGPRs).
#pragma unroll
        for (int i = 0; i < 16; ++i) {
            const int idx = tid + i * 256;
            const int t = idx >> 6, d = idx & 63;
            Vt[d * 64 + t] = Vg[headBase + (size_t)(kv0 + t) * Dh + d];
        }
        wait_async0();          // Q (first iter) + K tiles resident in LDS
        __syncthreads();

        // S = Q K^T (wave owns 16 query rows x 64 keys).
        v8f s[4];
#pragma unroll
        for (int tn = 0; tn < 4; ++tn) s[tn] = zero8();
#pragma unroll
        for (int kt = 0; kt < 2; ++kt) {
            const v16bf a = load_a(&Qs[(wave * 16 + ln) * 64 + kt * 32], hf);
#pragma unroll
            for (int tn = 0; tn < 4; ++tn) {
                const v16bf bb = load_b(&Ks[(tn * 16 + ln) * 64 + kt * 32], hf);
                s[tn] = WMMA_BF16(a, bb, s[tn]);
            }
        }

        // Scale + causal mask + online softmax.
        const int qbase = qt * 128 + wave * 16 + hf * 8;
        float sv[4][8];
#pragma unroll
        for (int tn = 0; tn < 4; ++tn) {
            const int kc = kv0 + tn * 16 + ln;
#pragma unroll
            for (int r = 0; r < 8; ++r) {
                float x = s[tn][r] * scale;
                if (kc > qbase + r) x = -1e30f;
                sv[tn][r] = x;
            }
        }
#pragma unroll
        for (int r = 0; r < 8; ++r) {
            float mx = sv[0][r];
#pragma unroll
            for (int tn = 1; tn < 4; ++tn) mx = fmaxf(mx, sv[tn][r]);
            mx = fmaxf(mx, __shfl_xor(mx, 1));
            mx = fmaxf(mx, __shfl_xor(mx, 2));
            mx = fmaxf(mx, __shfl_xor(mx, 4));
            mx = fmaxf(mx, __shfl_xor(mx, 8));
            const float mn = fmaxf(mI[r], mx);
            const float al = __expf(mI[r] - mn);
            mI[r] = mn;
            float rs = 0.f;
#pragma unroll
            for (int tn = 0; tn < 4; ++tn) {
                sv[tn][r] = __expf(sv[tn][r] - mn);
                rs += sv[tn][r];
            }
            rs += __shfl_xor(rs, 1);
            rs += __shfl_xor(rs, 2);
            rs += __shfl_xor(rs, 4);
            rs += __shfl_xor(rs, 8);
            lI[r] = lI[r] * al + rs;
#pragma unroll
            for (int tn = 0; tn < 4; ++tn) o[tn][r] *= al;
        }

        // P (C/D layout) -> LDS -> A layout for P@V.
        __bf16* Pw = &Ps[wave * 16 * 64];
#pragma unroll
        for (int tn = 0; tn < 4; ++tn)
#pragma unroll
            for (int r = 0; r < 8; ++r)
                Pw[(hf * 8 + r) * 64 + tn * 16 + ln] = (__bf16)sv[tn][r];
        __syncthreads();

        // O += P V
#pragma unroll
        for (int kt = 0; kt < 2; ++kt) {
            const v16bf a = load_a(&Pw[ln * 64 + kt * 32], hf);
#pragma unroll
            for (int tn = 0; tn < 4; ++tn) {
                const v16bf bb = load_b(&Vt[(tn * 16 + ln) * 64 + kt * 32], hf);
                o[tn] = WMMA_BF16(a, bb, o[tn]);
            }
        }
        __syncthreads();
    }

    // Normalize and write attn_out [B,T,C] bf16.
#pragma unroll
    for (int tn = 0; tn < 4; ++tn)
#pragma unroll
        for (int r = 0; r < 8; ++r) {
            const int q = qt * 128 + wave * 16 + hf * 8 + r;
            Og[((size_t)b * T + q) * C + h * Dh + tn * 16 + ln] =
                (__bf16)(o[tn][r] / lI[r]);
        }
}

// ---------------------------------------------------------------------------
// Kernel 3: out = attn_out(bf16) @ w_proj + b_proj  (M=8192, K=1024, N=1024)
// A tile staged via GLOBAL_LOAD_ASYNC_TO_LDS_B128.
// ---------------------------------------------------------------------------
__global__ __launch_bounds__(256) void proj_gemm(
    const __bf16* __restrict__ A, const float* __restrict__ W,
    const float* __restrict__ bias, float* __restrict__ Out)
{
    constexpr int Kd = C, N = C;
    __shared__ __bf16 As[128 * 32];   // [m][k]
    __shared__ __bf16 Bs[128 * 32];   // [n][k]

    const int tid  = threadIdx.x;
    const int lane = tid & 31, wave = tid >> 5;
    const int ln   = lane & 15, hf = lane >> 4;
    const int wm   = wave & 3, wn = wave >> 2;
    const int m0   = blockIdx.y * 128;
    const int n0   = blockIdx.x * 128;

    v8f acc[2][4];
#pragma unroll
    for (int mi = 0; mi < 2; ++mi)
#pragma unroll
        for (int ni = 0; ni < 4; ++ni) acc[mi][ni] = zero8();

    const int bn = tid & 127, bkh = (tid >> 7) * 16;

    for (int k0 = 0; k0 < Kd; k0 += 32) {
        // A tile: straight bf16 copy -> async to LDS.
#pragma unroll
        for (int i = 0; i < 2; ++i) {
            const int idx = tid + i * 256;
            const int row = idx >> 2, kg = (idx & 3) * 8;
            async_copy_b128(&As[row * 32 + kg],
                            A + (size_t)(m0 + row) * Kd + k0 + kg);
        }
        // W tile -> [n][k] bf16 (converts, stays on VGPR path).
#pragma unroll
        for (int g = 0; g < 4; ++g) {
            bf16x4 pk;
#pragma unroll
            for (int j = 0; j < 4; ++j)
                pk[j] = (__bf16)W[(size_t)(k0 + bkh + g * 4 + j) * N + n0 + bn];
            *(bf16x4*)(&Bs[bn * 32 + bkh + g * 4]) = pk;
        }
        if (k0 + 32 < Kd)
            __builtin_prefetch(W + (size_t)(k0 + 32 + bkh) * N + n0 + bn, 0, 1);
        wait_async0();
        __syncthreads();

        const v16bf a0 = load_a(&As[(wm * 32 +      ln) * 32], hf);
        const v16bf a1 = load_a(&As[(wm * 32 + 16 + ln) * 32], hf);
#pragma unroll
        for (int ni = 0; ni < 4; ++ni) {
            const v16bf b = load_b(&Bs[(wn * 64 + ni * 16 + ln) * 32], hf);
            acc[0][ni] = WMMA_BF16(a0, b, acc[0][ni]);
            acc[1][ni] = WMMA_BF16(a1, b, acc[1][ni]);
        }
        __syncthreads();
    }

#pragma unroll
    for (int ni = 0; ni < 4; ++ni) {
        const int j = n0 + wn * 64 + ni * 16 + ln;
        const float bj = bias[j];
#pragma unroll
        for (int mi = 0; mi < 2; ++mi)
#pragma unroll
            for (int r = 0; r < 8; ++r) {
                const int row = m0 + wm * 32 + mi * 16 + hf * 8 + r;
                Out[(size_t)row * N + j] = acc[mi][ni][r] + bj;
            }
    }
}

// ---------------------------------------------------------------------------
extern "C" void kernel_launch(void* const* d_in, const int* in_sizes, int n_in,
                              void* d_out, int out_size, void* d_ws, size_t ws_size,
                              hipStream_t stream) {
    const float* x      = (const float*)d_in[0];
    const float* w_qkv  = (const float*)d_in[1];
    const float* b_qkv  = (const float*)d_in[2];
    const float* w_proj = (const float*)d_in[3];
    const float* b_proj = (const float*)d_in[4];
    float* out = (float*)d_out;

    const size_t headElems = (size_t)Bsz * T * C;   // 8,388,608 per tensor
    __bf16* Q  = (__bf16*)d_ws;
    __bf16* K  = Q + headElems;
    __bf16* V  = K + headElems;
    __bf16* AO = V + headElems;                     // 64 MB total bf16

    // qkv: M=8192 (64 tiles), N=3072 (24 tiles)
    qkv_gemm<<<dim3(24, 64), 256, 0, stream>>>(x, w_qkv, b_qkv, Q, K, V);
    // attention: 16 query tiles x 16 heads x 4 batches
    flash_attn<<<dim3(16, 16, 4), 256, 0, stream>>>(Q, K, V, AO);
    // proj: M=8192 (64 tiles), N=1024 (8 tiles)
    proj_gemm<<<dim3(8, 64), 256, 0, stream>>>(AO, w_proj, b_proj, out);
}
